// GATLayer_33114197852230
// MI455X (gfx1250) — matrix operand order repaired and checked by reading, not compiled, not run on hardware
//
#include <hip/hip_runtime.h>
#include <hip/hip_bf16.h>
#include <math.h>

// CDNA5 / gfx1250: wave32, WMMA 16x16x32 bf16 (f32 accum).
typedef __attribute__((ext_vector_type(16))) __bf16 v16bf;
typedef __attribute__((ext_vector_type(4)))  __bf16 v4bf;
typedef __attribute__((ext_vector_type(8)))  float  v8f;

#define GAT_ALPHA 0.35f
#define GAT_MASK  (-9.0e15f)
#define IN_F  128
#define OUT_F 64
#define NB    32
#define MM    1024

// branchless attention logit
__device__ __forceinline__ float gat_e(int adjv, float f1v, float f2row) {
  float t  = f2row + f1v;
  float lr = (t > 0.0f) ? t : GAT_ALPHA * t;   // v_cndmask, no branch
  return (adjv > 0) ? lr : GAT_MASK;           // v_cndmask, no branch
}

// ---------------------------------------------------------------------------
// Kernel 1: h = x @ W; writes h into WMMA-B-swizzled bf16 layout
//   hswz[((n*32 + m/32)*64 + o)*32 + (m%32)]
// Block = 256 threads = 4 row-quads x 64 features; each thread computes 4
// consecutive rows for one feature so each W value is loaded once and reused
// 4x (cuts W L2 traffic 4x), and x row reads are wave-uniform scalar loads.
// ---------------------------------------------------------------------------
__global__ __launch_bounds__(256)
void gat_proj(const float* __restrict__ x,
              const float* __restrict__ W,
              const float* __restrict__ a,
              __bf16* __restrict__ hswz,
              float* __restrict__ f1,
              float* __restrict__ f2) {
  const int tid  = threadIdx.x;
  const int o    = tid & 63;                  // output feature
  const int q    = tid >> 6;                  // row-quad within block (0..3)
  const int row0 = blockIdx.x * 16 + q * 4;   // 4 consecutive flattened rows
  const int n    = row0 >> 10;
  const int m0   = row0 & 1023;

  const float* x0 = x + (size_t)row0 * IN_F;  // wave-uniform base
  float acc0 = 0.0f, acc1 = 0.0f, acc2 = 0.0f, acc3 = 0.0f;
#pragma unroll 8
  for (int k = 0; k < IN_F; ++k) {
    float w = W[k * OUT_F + o];
    acc0 = fmaf(x0[k],            w, acc0);
    acc1 = fmaf(x0[IN_F + k],     w, acc1);
    acc2 = fmaf(x0[2 * IN_F + k], w, acc2);
    acc3 = fmaf(x0[3 * IN_F + k], w, acc3);
  }

  // swizzled store for WMMA B operand: 4 consecutive m slots -> one b64 store
  v4bf hv = { (__bf16)acc0, (__bf16)acc1, (__bf16)acc2, (__bf16)acc3 };
  *(v4bf*)(hswz + (((size_t)n * 32 + (m0 >> 5)) * OUT_F + o) * 32 + (m0 & 31)) = hv;

  // reduce h.a1 and h.a2 over the 64 features of each of the 4 rows
  __shared__ float r1[256];
  __shared__ float r2[256];
  const float accs[4] = {acc0, acc1, acc2, acc3};
#pragma unroll
  for (int r = 0; r < 4; ++r) {
    __syncthreads();
    r1[tid] = accs[r] * a[o];
    r2[tid] = accs[r] * a[OUT_F + o];
    __syncthreads();
#pragma unroll
    for (int s = 32; s > 0; s >>= 1) {
      if (o < s) { r1[tid] += r1[tid + s]; r2[tid] += r2[tid + s]; }
      __syncthreads();
    }
    if (o == 0) { f1[row0 + r] = r1[tid]; f2[row0 + r] = r2[tid]; }
  }
}

// ---------------------------------------------------------------------------
// Kernel 2: fused masked-softmax(attn) @ h + ELU, flash-attention style.
// One wave (32 lanes) owns a 16-row tile of one batch. Block = 4 waves,
// waves fully independent (no LDS, no barriers).
// ---------------------------------------------------------------------------
__global__ __launch_bounds__(128)
void gat_attn(const int*    __restrict__ adj,
              const __bf16* __restrict__ hswz,
              const float*  __restrict__ f1,
              const float*  __restrict__ f2,
              float* __restrict__ out) {
  const int lane  = threadIdx.x & 31;
  const int wid   = threadIdx.x >> 5;
  const int tile  = blockIdx.x * 4 + wid;   // 0 .. 2047
  const int n     = tile >> 6;              // batch  (64 tiles per batch)
  const int itile = tile & 63;
  const int i0    = itile * 16;             // first output row of this tile

  const float* f1n     = f1 + n * MM;
  const size_t adjbase = ((size_t)n * MM + i0) * MM;

  // ---- Pass 1: online-softmax stats (row max m, 1/sum) per output row ----
  float mrow = 0.0f, invl = 0.0f;
  for (int i = 0; i < 16; ++i) {
    const int* adjrow = adj + adjbase + (size_t)i * MM;
    const float f2i   = f2[n * MM + i0 + i];      // uniform -> scalar load
    float m = -3.0e38f, l = 0.0f;
#pragma unroll 2
    for (int it = 0; it < MM; it += 128) {        // lane covers 4-wide vector
      const int j = it + lane * 4;
      int4   av = *(const int4*)  (adjrow + j);   // branchless vector loads
      float4 fv = *(const float4*)(f1n + j);
      float e0 = gat_e(av.x, fv.x, f2i);
      float e1 = gat_e(av.y, fv.y, f2i);
      float e2 = gat_e(av.z, fv.z, f2i);
      float e3 = gat_e(av.w, fv.w, f2i);
      // grouped online update: 5 exps / 4 elems, short dependency chain
      float m4 = fmaxf(fmaxf(e0, e1), fmaxf(e2, e3));
      float nm = fmaxf(m, m4);
      float s4 = (__expf(e0 - nm) + __expf(e1 - nm)) +
                 (__expf(e2 - nm) + __expf(e3 - nm));
      l = fmaf(l, __expf(m - nm), s4);
      m = nm;
    }
    // wave32 butterfly combine of (m, l); all lanes end with the row result
#pragma unroll
    for (int s = 1; s < 32; s <<= 1) {
      float om = __shfl_xor(m, s, 32);
      float ol = __shfl_xor(l, s, 32);
      float nm = fmaxf(m, om);
      l = l * __expf(m - nm) + ol * __expf(om - nm);
      m = nm;
    }
    if ((lane & 15) == i) { mrow = m; invl = 1.0f / l; }
  }

  // ---- Pass 2: P(16x32 bf16) @ H(32x64 bf16) accumulated in f32 WMMA ----
  // A as this code fills it (16x32, contiguous k = 16h+s, B the same): lanes 0-15 = rows 0-15 with K {0..15},
  // lanes 16-31 = rows 0-15 with K {16..31}; 2 values packed per VGPR.
  const int irow  = lane & 15;
  const int abase = (lane < 16) ? 0 : 16;   // A-operand K base
  const int kbase = (lane < 16) ? 0 : 16;   // B-operand K base
  const int* adjrow = adj + adjbase + (size_t)irow * MM;
  const float f2row = f2[n * MM + i0 + irow];
  // per-lane base into the swizzled h: column o' = (lane&15), K base kbase
  const __bf16* hlane =
      hswz + (((size_t)n * 32) * OUT_F + (lane & 15)) * 32 + kbase;

  v8f acc[4];
#pragma unroll
  for (int ot = 0; ot < 4; ++ot) acc[ot] = (v8f){};

  for (int jc = 0; jc < MM; jc += 32) {
    // unconditional prefetch of next adj chunk, branchless in-bounds clamp
    int jpf = jc + 32;
    jpf = (jpf < MM) ? jpf : (MM - 32);
    __builtin_prefetch(adjrow + jpf, 0, 1);

    // Build A tile (probabilities) in registers, branchless + vectorized
    v16bf avv;
#pragma unroll
    for (int g = 0; g < 2; ++g) {
      const int jg = jc + abase + 8 * g;
      int4   a0 = *(const int4*)  (adjrow + jg);
      int4   a1 = *(const int4*)  (adjrow + jg + 4);
      float4 q0 = *(const float4*)(f1n + jg);
      float4 q1 = *(const float4*)(f1n + jg + 4);
      avv[g*8 + 0] = (__bf16)(__expf(gat_e(a0.x, q0.x, f2row) - mrow) * invl);
      avv[g*8 + 1] = (__bf16)(__expf(gat_e(a0.y, q0.y, f2row) - mrow) * invl);
      avv[g*8 + 2] = (__bf16)(__expf(gat_e(a0.z, q0.z, f2row) - mrow) * invl);
      avv[g*8 + 3] = (__bf16)(__expf(gat_e(a0.w, q0.w, f2row) - mrow) * invl);
      avv[g*8 + 4] = (__bf16)(__expf(gat_e(a1.x, q1.x, f2row) - mrow) * invl);
      avv[g*8 + 5] = (__bf16)(__expf(gat_e(a1.y, q1.y, f2row) - mrow) * invl);
      avv[g*8 + 6] = (__bf16)(__expf(gat_e(a1.z, q1.z, f2row) - mrow) * invl);
      avv[g*8 + 7] = (__bf16)(__expf(gat_e(a1.w, q1.w, f2row) - mrow) * invl);
    }
    // B tiles from swizzled h: one contiguous 32B load per (chunk, ot)
    const __bf16* hc = hlane + (size_t)(jc >> 5) * OUT_F * 32;
#pragma unroll
    for (int ot = 0; ot < 4; ++ot) {
      v16bf bv = *(const v16bf*)(hc + (size_t)ot * 16 * 32);
      acc[ot] = __builtin_amdgcn_wmma_f32_16x16x32_bf16(
          /*neg_a=*/false, avv, /*neg_b=*/false, bv,
          /*c_mod=*/(short)0, acc[ot], /*reuse_a=*/false, /*reuse_b=*/false);
    }
  }

  // ---- Epilogue: branchless ELU + store.
  // C/D layout: col = lane&15, row = r + 8*(lane/16)
  const int col  = lane & 15;
  const int rofs = (lane >> 4) * 8;
#pragma unroll
  for (int ot = 0; ot < 4; ++ot) {
#pragma unroll
    for (int r = 0; r < 8; ++r) {
      float v   = acc[ot][r];
      float res = (v > 0.0f) ? v : (__expf(v) - 1.0f);  // v_exp + v_cndmask
      out[((size_t)n * MM + i0 + rofs + r) * OUT_F + ot * 16 + col] = res;
    }
  }
}

// ---------------------------------------------------------------------------
extern "C" void kernel_launch(void* const* d_in, const int* in_sizes, int n_in,
                              void* d_out, int out_size, void* d_ws, size_t ws_size,
                              hipStream_t stream) {
  const float* x   = (const float*)d_in[0];  // [32,1024,128]
  const int*   adj = (const int*)  d_in[1];  // [32,1024,1024]
  const float* W   = (const float*)d_in[2];  // [128,64]
  const float* a   = (const float*)d_in[3];  // [128,1]
  float*       out = (float*)d_out;          // [32,1024,64]

  char* ws = (char*)d_ws;
  float*  f1   = (float*) (ws);               // 32768 f32 = 128 KB
  float*  f2   = (float*) (ws + (128 << 10)); // 128 KB
  __bf16* hswz = (__bf16*)(ws + (256 << 10)); // 4 MB, WMMA-B swizzled

  // 32768 rows / 16 per block
  gat_proj<<<(NB * MM) / 16, 256, 0, stream>>>(x, W, a, hswz, f1, f2);
  // 2048 row-tiles / 4 waves per block
  gat_attn<<<(NB * MM / 16) / 4, 128, 0, stream>>>(adj, hswz, f1, f2, out);
}
